// AdaptiveCross_74577812128432
// MI455X (gfx1250) — compile-verified, matrix operand order checked by reading, not run on hardware
//
#include <hip/hip_runtime.h>
#include <hip/hip_bf16.h>
#include <math.h>

// ---------------------------------------------------------------------------
// CDNA5 (gfx1250) implementation of the AdaptiveCross forward pass.
// All GEMMs lower to v_wmma_f32_16x16x32_f16; one wave32 computes a 32x32
// output tile (4 wmma / K-step). K is always a multiple of 32 (branch-free
// inner loop); all addressing is uniform-base + 32-bit lane offset so the
// backend can use scalar-base (saddr) global loads.
// ---------------------------------------------------------------------------

typedef __attribute__((ext_vector_type(16))) _Float16 v16h;
typedef __attribute__((ext_vector_type(8)))  float    v8f;

#define BNEPS 1e-5f

// Y[b,m,n] = alpha * sum_k A[b,m,k] * X[b,k,n] (+ bias[m]) (+ bias2[b,m,n>>7])
// MODE 0: X strided; MODE 1: X = dis_fea[b,k,j]-temp_fea[b,k,i] (n=i*128+j);
// MODE 2: X = dw[b,k,n]*temp_fea[b,k,i].
struct GemmDesc {
  const float* A; long sAb, sAm, sAk;
  const float* B; long sBb, sBk, sBn;
  const float* bias;
  const float* bias2;      // per (b,m,i) additive term (mode 2), else nullptr
  float alpha;
  float* Y; long sYb, sYm, sYn;
  int M, N, K, batch;
  const float* aux0; const float* aux1;
};

template <int MODE, bool AC>
__global__ void __launch_bounds__(32) wmma_gemm_kernel(GemmDesc d) {
  const int n0 = blockIdx.x << 5;
  const int m0 = blockIdx.y << 5;
  const int b  = blockIdx.z;
  const int lane = threadIdx.x;
  const int hs = lane >> 4;   // K half select
  const int lr = lane & 15;   // row (A) / column (B)

  // Uniform per-batch bases (scalar regs); per-lane offsets are 32-bit.
  const float* Ab = d.A + (long)b * d.sAb;
  const float* Bb = (MODE == 0) ? d.B + (long)b * d.sBb : nullptr;
  const float* X0 = (MODE == 1) ? d.aux0 + (long)b * 16384
                  : (MODE == 2) ? d.aux0 + (long)b * 2097152 : nullptr;
  const float* T0 = (MODE != 0) ? d.aux1 + (long)b * 16384 : nullptr;
  const int sBk = (int)d.sBk, sBn = (int)d.sBn;
  const int sAm = (int)d.sAm, sAk = (int)d.sAk;

  int nOff[2], aOff[2], nfull[2], iOff[2];
#pragma unroll
  for (int t = 0; t < 2; ++t) {
    int n = n0 + t * 16 + lr;
    nfull[t] = n;
    if (MODE == 0) nOff[t] = n * sBn;
    else           iOff[t] = (MODE == 1) ? (n & 127) : 0;  // j for mode1
    aOff[t] = (m0 + t * 16 + lr) * sAm;
  }
  int ti0[2];
#pragma unroll
  for (int t = 0; t < 2; ++t) ti0[t] = (nfull[t] >> 7);     // i index

  v8f acc[2][2] = {};
  const int ksteps = d.K >> 5;
  for (int kt = 0; kt < ksteps; ++kt) {
    const int kbase = kt << 5;
    v16h a[2], bf[2];
#pragma unroll
    for (int t = 0; t < 2; ++t) {
      if (AC) {
        // lane fragment = two contiguous 8-float runs -> 4x b128 loads
        const float* ap = Ab + aOff[t] + kbase + hs * 8;
        const float4 x0 = *(const float4*)(ap);
        const float4 x1 = *(const float4*)(ap + 4);
        const float4 x2 = *(const float4*)(ap + 16);
        const float4 x3 = *(const float4*)(ap + 20);
        a[t][0]  = (_Float16)x0.x; a[t][1]  = (_Float16)x0.y;
        a[t][2]  = (_Float16)x0.z; a[t][3]  = (_Float16)x0.w;
        a[t][4]  = (_Float16)x1.x; a[t][5]  = (_Float16)x1.y;
        a[t][6]  = (_Float16)x1.z; a[t][7]  = (_Float16)x1.w;
        a[t][8]  = (_Float16)x2.x; a[t][9]  = (_Float16)x2.y;
        a[t][10] = (_Float16)x2.z; a[t][11] = (_Float16)x2.w;
        a[t][12] = (_Float16)x3.x; a[t][13] = (_Float16)x3.y;
        a[t][14] = (_Float16)x3.z; a[t][15] = (_Float16)x3.w;
      } else {
#pragma unroll
        for (int e = 0; e < 16; ++e) {
          int k = kbase + ((e < 8) ? 0 : 16) + hs * 8 + (e & 7);
          a[t][e] = (_Float16)Ab[aOff[t] + k * sAk];
        }
      }
#pragma unroll
      for (int e = 0; e < 16; ++e) {
        int k = kbase + ((e < 8) ? 0 : 16) + hs * 8 + (e & 7);
        float x;
        if (MODE == 0) {
          x = Bb[nOff[t] + k * sBk];
        } else if (MODE == 1) {
          x = X0[k * 128 + iOff[t]] - T0[k * 128 + ti0[t]];
        } else {
          x = X0[k * 16384 + nfull[t]] * T0[k * 128 + ti0[t]];
        }
        bf[t][e] = (_Float16)x;
      }
    }
    acc[0][0] = __builtin_amdgcn_wmma_f32_16x16x32_f16(false, a[0], false, bf[0], (short)0, acc[0][0], false, false);
    acc[0][1] = __builtin_amdgcn_wmma_f32_16x16x32_f16(false, a[0], false, bf[1], (short)0, acc[0][1], false, false);
    acc[1][0] = __builtin_amdgcn_wmma_f32_16x16x32_f16(false, a[1], false, bf[0], (short)0, acc[1][0], false, false);
    acc[1][1] = __builtin_amdgcn_wmma_f32_16x16x32_f16(false, a[1], false, bf[1], (short)0, acc[1][1], false, false);
  }

  float* Yb = d.Y + (long)b * d.sYb;
  const float* B2 = d.bias2 ? d.bias2 + (long)b * d.M * 128 : nullptr;
  const int sYm = (int)d.sYm, sYn = (int)d.sYn;
#pragma unroll
  for (int ti = 0; ti < 2; ++ti) {
#pragma unroll
    for (int tj = 0; tj < 2; ++tj) {
#pragma unroll
      for (int v = 0; v < 8; ++v) {
        int m = m0 + ti * 16 + v + hs * 8;
        int n = n0 + tj * 16 + lr;
        float y = d.alpha * acc[ti][tj][v];
        if (d.bias) y += d.bias[m];
        if (B2)     y += B2[m * 128 + (n >> 7)];
        Yb[m * sYm + n * sYn] = y;
      }
    }
  }
}

// ---------------------------------------------------------------------------
// BatchNorm (train mode) + activations. act: 0 none,1 relu,2 leaky(.2),3 sigmoid
// ---------------------------------------------------------------------------
__global__ void bn_stats_kernel(const float* __restrict__ Y, float* mean, float* var,
                                int C, long L, int B) {
  int c = blockIdx.x;
  __shared__ float ss[256], sq[256];
  float s = 0.f, q = 0.f;
  for (int bb = 0; bb < B; ++bb) {
    const float* p = Y + ((long)bb * C + c) * L;
    for (long l = threadIdx.x; l < L; l += 256) {
      float x = p[l];
      s += x; q += x * x;
    }
  }
  ss[threadIdx.x] = s; sq[threadIdx.x] = q;
  __syncthreads();
  for (int off = 128; off > 0; off >>= 1) {
    if (threadIdx.x < off) { ss[threadIdx.x] += ss[threadIdx.x + off];
                             sq[threadIdx.x] += sq[threadIdx.x + off]; }
    __syncthreads();
  }
  if (threadIdx.x == 0) {
    float per = (float)((long)B * L);
    float m = ss[0] / per;
    mean[c] = m;
    var[c]  = sq[0] / per - m * m;
  }
}

__global__ void bn_act_kernel(float* Y, const float* mean, const float* var,
                              const float* g, const float* bt, int act,
                              int C, long L, long total) {
  long i = (long)blockIdx.x * blockDim.x + threadIdx.x;
  if (i >= total) return;
  int c = (int)((i / L) % C);
  float x = Y[i];
  if (mean) x = g[c] * (x - mean[c]) * rsqrtf(var[c] + BNEPS) + bt[c];
  if      (act == 1) x = fmaxf(x, 0.f);
  else if (act == 2) x = (x > 0.f) ? x : 0.2f * x;
  else if (act == 3) x = 1.f / (1.f + expf(-x));
  Y[i] = x;
}

// out[b,d2,d1] = in[b,d1,d2]
__global__ void transpose12_kernel(float* out, const float* in, int B, int D1, int D2) {
  long total = (long)B * D1 * D2;
  long i = (long)blockIdx.x * blockDim.x + threadIdx.x;
  if (i >= total) return;
  int d2 = (int)(i % D2);
  long t = i / D2;
  int d1 = (int)(t % D1);
  int b  = (int)(t / D1);
  out[((long)b * D2 + d2) * D1 + d1] = in[i];
}

// slice/pad weights: out[m*Kout+k] = (k+off < Kin) ? in[m*Kin+k+off] : 0
__global__ void slice_pad_w_kernel(float* out, const float* in,
                                   int M, int Kin, int Kout, int off) {
  int i = blockIdx.x * blockDim.x + threadIdx.x;
  if (i >= M * Kout) return;
  int k = i % Kout, m = i / Kout;
  out[i] = (k + off < Kin) ? in[(long)m * Kin + k + off] : 0.f;
}

// xyz part of dis_mlp[0] collapses to a per-(b,m,i) bias: W[:,0:3] @ temp_xyz + b
__global__ void xyzterm_kernel(float* term, const float* W, const float* bias,
                               const float* xyz) {
  int t = blockIdx.x * blockDim.x + threadIdx.x;  // (b,m,i) : 16*128*128
  if (t >= 16 * 128 * 128) return;
  int i = t & 127; int bm = t >> 7; int m = bm & 127; int b = bm >> 7;
  const float* w = W + (long)m * 131;
  const float* x = xyz + (long)b * 3 * 128 + i;
  term[t] = bias[m] + w[0] * x[0] + w[1] * x[128] + w[2] * x[256];
}

// K=3 positional-encoding layer (3 FMAs/point; not worth a WMMA)
__global__ void posenc_l1_kernel(float* Y, const float* W, const float* bias,
                                 const float* xyz) {
  long t = (long)blockIdx.x * blockDim.x + threadIdx.x;  // (b,m,l): 16*128*256
  if (t >= 16L * 128 * 256) return;
  int l = (int)(t & 255); long bm = t >> 8;
  int m = (int)(bm & 127); int b = (int)(bm >> 7);
  const float* w = W + (long)m * 3;
  const float* x = xyz + (long)b * 3 * 256 + l;
  Y[t] = bias[m] + w[0] * x[0] + w[1] * x[256] + w[2] * x[512];
}

// s[b,c] = sum_p con_fea[b,c,p] ; nrm[b,c] = ||con_fea[b,c,:]||
__global__ void confuse_stats_kernel(const float* conf, float* s, float* nrm, int B) {
  int i = blockIdx.x * blockDim.x + threadIdx.x;
  if (i >= B * 128) return;
  const float* row = conf + (long)i * 128;
  float su = 0.f, sq = 0.f;
  for (int p = 0; p < 128; ++p) { float x = row[p]; su += x; sq += x * x; }
  s[i] = su; nrm[i] = sqrtf(sq);
}

// per (b,t): cosine-sim over channels, argmax (first max), max, gather from txf
__global__ void confuse_sim_kernel(const float* temp_fea, const float* temp_xyz,
                                   const float* s, const float* nrm,
                                   float* maxcla, float* gathered, int B) {
  int i = blockIdx.x * blockDim.x + threadIdx.x;
  if (i >= B * 128) return;
  int b = i >> 7, t = i & 127;
  const float rootn2 = 11.313708498984761f;  // sqrt(128)
  float best = -1e30f; int bi = 0;
  for (int c = 0; c < 128; ++c) {
    float tf  = temp_fea[((long)b * 128 + c) * 128 + t];
    float den = fmaxf(fabsf(tf) * rootn2 * nrm[b * 128 + c], 1e-8f);
    float sim = tf * s[b * 128 + c] / den;
    if (sim > best) { best = sim; bi = c; }
  }
  maxcla[i] = best;
  gathered[i] = (bi < 3) ? temp_xyz[((long)b * 3 + bi) * 128 + t]
                         : temp_fea[((long)b * 128 + (bi - 3)) * 128 + t];
}

// fusion X (B,288,128): c0=max_cla, c1..131=gathered, c132..259=con_fea, pad 0
__global__ void build_fusion_kernel(float* X, const float* maxcla, const float* gathered,
                                    const float* conf, int B) {
  long total = (long)B * 288 * 128;
  long i = (long)blockIdx.x * blockDim.x + threadIdx.x;
  if (i >= total) return;
  int t = (int)(i % 128);
  long r = i / 128;
  int c = (int)(r % 288);
  int b = (int)(r / 288);
  float v;
  if (c >= 260)     v = 0.f;
  else if (c == 0)  v = maxcla[b * 128 + t];
  else if (c < 132) v = gathered[b * 128 + t];
  else              v = conf[((long)b * 128 + (c - 132)) * 128 + t];
  X[i] = v;
}

// dis_feature[b,c,j] = max_i in[(b*128+c)*16384 + i*128 + j]
__global__ void maxpool_i_kernel(float* out, const float* in, int B) {
  int i = blockIdx.x * blockDim.x + threadIdx.x;
  if (i >= B * 128 * 128) return;
  int j = i & 127;
  long bc = (long)(i >> 7);
  const float* p = in + bc * 16384 + j;
  float m = -1e30f;
  for (int ii = 0; ii < 128; ++ii) m = fmaxf(m, p[ii * 128]);
  out[i] = m;
}

// fused (B,128,256): l<128 from a, else from b (both (B,128,128))
__global__ void concat_kernel(float* out, const float* a, const float* b, int B) {
  long total = (long)B * 128 * 256;
  long i = (long)blockIdx.x * blockDim.x + threadIdx.x;
  if (i >= total) return;
  int l = (int)(i % 256);
  long bc = i / 256;
  out[i] = (l < 128) ? a[bc * 128 + l] : b[bc * 128 + (l - 128)];
}

// op 0: x+y ; op 1: x-y
__global__ void axpy_kernel(float* out, const float* x, const float* y, int op, long total) {
  long i = (long)blockIdx.x * blockDim.x + threadIdx.x;
  if (i >= total) return;
  out[i] = (op == 0) ? (x[i] + y[i]) : (x[i] - y[i]);
}

// L2-normalize over channel dim of (16,128,256); one block per (b,l)
__global__ void l2norm_ch_kernel(float* Y) {
  __shared__ float sh[128];
  int bl = blockIdx.x;
  int b = bl >> 8, l = bl & 255;
  int c = threadIdx.x;
  long idx = ((long)b * 128 + c) * 256 + l;
  float x = Y[idx];
  sh[c] = x * x;
  __syncthreads();
  for (int off = 64; off > 0; off >>= 1) {
    if (c < off) sh[c] += sh[c + off];
    __syncthreads();
  }
  float nrm = fmaxf(sqrtf(sh[0]), 1e-12f);
  Y[idx] = x / nrm;
}

// softmax over rows of 256 (logits (B,256,256)); one block per row
__global__ void softmax_kernel(float* L) {
  __shared__ float sh[256];
  long row = blockIdx.x;
  int t = threadIdx.x;
  float x = L[row * 256 + t];
  sh[t] = x; __syncthreads();
  for (int off = 128; off > 0; off >>= 1) {
    if (t < off) sh[t] = fmaxf(sh[t], sh[t + off]);
    __syncthreads();
  }
  float mx = sh[0]; __syncthreads();
  float e = expf(x - mx);
  sh[t] = e; __syncthreads();
  for (int off = 128; off > 0; off >>= 1) {
    if (t < off) sh[t] += sh[t + off];
    __syncthreads();
  }
  L[row * 256 + t] = e / sh[0];
}

// out[b,c,:] = InstanceNorm_L(base[b,c,:] + att[b,c,:]); one block per (b,c)
__global__ void add_inorm_kernel(float* out, const float* base, const float* att) {
  __shared__ float sh[256];
  long bc = blockIdx.x;
  int l = threadIdx.x;
  float x = base[bc * 256 + l] + att[bc * 256 + l];
  sh[l] = x; __syncthreads();
  for (int off = 128; off > 0; off >>= 1) {
    if (l < off) sh[l] += sh[l + off];
    __syncthreads();
  }
  float m = sh[0] * (1.f / 256.f); __syncthreads();
  float d0 = x - m;
  sh[l] = d0 * d0; __syncthreads();
  for (int off = 128; off > 0; off >>= 1) {
    if (l < off) sh[l] += sh[l + off];
    __syncthreads();
  }
  float v = sh[0] * (1.f / 256.f);
  out[bc * 256 + l] = d0 * rsqrtf(v + BNEPS);
}

// ---------------------------------------------------------------------------
// Host-side helpers
// ---------------------------------------------------------------------------
static void launch_gemm(const GemmDesc& d, hipStream_t st, int mode) {
  dim3 grid((unsigned)(d.N >> 5), (unsigned)(d.M >> 5), (unsigned)d.batch);
  dim3 blk(32);
  bool ac = (d.sAk == 1);
  if (mode == 0) {
    if (ac) wmma_gemm_kernel<0, true ><<<grid, blk, 0, st>>>(d);
    else    wmma_gemm_kernel<0, false><<<grid, blk, 0, st>>>(d);
  } else if (mode == 1) {
    wmma_gemm_kernel<1, true><<<grid, blk, 0, st>>>(d);
  } else {
    wmma_gemm_kernel<2, true><<<grid, blk, 0, st>>>(d);
  }
}

static void bn_block(hipStream_t st, float* Y, int M, long N,
                     const float* g, const float* bt, int act,
                     float* mean, float* var) {
  long total = 16L * M * N;
  bn_stats_kernel<<<M, 256, 0, st>>>(Y, mean, var, M, N, 16);
  bn_act_kernel<<<(int)((total + 255) / 256), 256, 0, st>>>(
      Y, mean, var, g, bt, act, M, N, total);
}

// Conv-style layer: Y (16,M,N) contiguous = W(MxK) * X + bias, then BN/act.
static void conv_layer(hipStream_t st,
                       const float* W, const float* bias,
                       const float* g, const float* bt,
                       const float* X, long sBb, long sBk, long sBn,
                       float* Y, int M, int N, int K, int act,
                       float* mean, float* var,
                       int mode = 0, const float* a0 = nullptr,
                       const float* a1 = nullptr, const float* bias2 = nullptr) {
  GemmDesc d{};
  d.A = W; d.sAb = 0; d.sAm = K; d.sAk = 1;
  d.B = X; d.sBb = sBb; d.sBk = sBk; d.sBn = sBn;
  d.bias = bias; d.bias2 = bias2; d.alpha = 1.f;
  d.Y = Y; d.sYb = (long)M * N; d.sYm = N; d.sYn = 1;
  d.M = M; d.N = N; d.K = K; d.batch = 16;
  d.aux0 = a0; d.aux1 = a1;
  launch_gemm(d, st, mode);
  if (g) {
    bn_block(st, Y, M, (long)N, g, bt, act, mean, var);
  } else if (act) {
    long total = 16L * M * N;
    bn_act_kernel<<<(int)((total + 255) / 256), 256, 0, st>>>(
        Y, nullptr, nullptr, nullptr, nullptr, act, M, (long)N, total);
  }
}

// Shared-WK projection + L2 normalize: Y[b,o,l] = l2n_o(WK q + bk)
static void att_proj(hipStream_t st, const float* wk, const float* bk,
                     const float* X, float* Y) {
  conv_layer(st, wk, bk, nullptr, nullptr, X, 32768, 256, 1, Y, 128, 256, 128, 0,
             nullptr, nullptr);
  l2norm_ch_kernel<<<16 * 256, 128, 0, st>>>(Y);
}

// logits = 30 * wq^T wk2 (per batch), softmax, out = inorm(base + A @ V)
static void att_core(hipStream_t st, const float* wq, const float* wk2,
                     const float* v, const float* base, float* out,
                     float* logits, float* atto) {
  GemmDesc d{};
  d.A = wq;   d.sAb = 32768; d.sAm = 1;   d.sAk = 256;   // strided A path
  d.B = wk2;  d.sBb = 32768; d.sBk = 256; d.sBn = 1;
  d.bias = nullptr; d.bias2 = nullptr; d.alpha = 30.f;
  d.Y = logits; d.sYb = 65536; d.sYm = 256; d.sYn = 1;
  d.M = 256; d.N = 256; d.K = 128; d.batch = 16;
  launch_gemm(d, st, 0);
  softmax_kernel<<<16 * 256, 256, 0, st>>>(logits);
  GemmDesc e{};
  e.A = v;      e.sAb = 32768; e.sAm = 256; e.sAk = 1;    // A[b,c,k] = v[b,c,k]
  e.B = logits; e.sBb = 65536; e.sBk = 1;   e.sBn = 256;  // B[b,k,q] = a[b,q,k]
  e.bias = nullptr; e.bias2 = nullptr; e.alpha = 1.f;
  e.Y = atto; e.sYb = 32768; e.sYm = 256; e.sYn = 1;
  e.M = 128; e.N = 256; e.K = 256; e.batch = 16;
  launch_gemm(e, st, 0);
  add_inorm_kernel<<<16 * 128, 256, 0, st>>>(out, base, atto);
}

// ---------------------------------------------------------------------------
extern "C" void kernel_launch(void* const* d_in, const int* in_sizes, int n_in,
                              void* d_out, int out_size, void* d_ws, size_t ws_size,
                              hipStream_t stream) {
  (void)in_sizes; (void)out_size; (void)ws_size;

  // Param pointers follow jax tree-flatten order (dict keys sorted; b<bt<g<w).
  const float* P[106] = {nullptr};
  for (int i = 0; i < 106 && i < n_in; ++i) P[i] = (const float*)d_in[i];

  const float* search_fea = P[0];  // (16,128,256)
  const float* temp_fea   = P[1];  // (16,128,128)
  const float* temp_xyz   = P[2];  // (16,3,128)
  const float* search_xyz = P[3];  // (16,3,256)
  float* out = (float*)d_out;      // (16,128,256)

  // ----- workspace bump allocator (floats) -----
  float* ws = (float*)d_ws;
  size_t off = 0;
  auto alloc = [&](size_t n) -> float* {
    float* p = ws + off;
    off += (n + 63) & ~(size_t)63;
    return p;
  };
  float* big0     = alloc(16L * 128 * 16384);  // 128 MB (L2-resident)
  float* big1     = alloc(16L * 128 * 16384);  // 128 MB
  float* mean     = alloc(256);
  float* var      = alloc(256);
  float* t0       = alloc(16L * 256 * 128);
  float* t1       = alloc(16L * 256 * 128);
  float* tr4      = alloc(16L * 128 * 128);
  float* confea   = alloc(16L * 128 * 128);
  float* disfea   = alloc(16L * 128 * 128);
  float* sbuf     = alloc(16 * 128);
  float* nrmb     = alloc(16 * 128);
  float* maxcla   = alloc(16 * 128);
  float* gath     = alloc(16 * 128);
  float* fusX     = alloc(16L * 288 * 128);    // K padded 260 -> 288
  float* wp288    = alloc(128L * 288);
  float* wdis     = alloc(128L * 128);
  float* xyzt     = alloc(16L * 128 * 128);
  float* cm0      = alloc(16L * 128 * 128);
  float* cm1      = alloc(16L * 128 * 128);
  float* disfeat  = alloc(16L * 128 * 128);
  float* fusedcat = alloc(16L * 128 * 256);
  float* fa       = alloc(16L * 128 * 256);
  float* fb       = alloc(16L * 128 * 256);
  float* fusedout = alloc(16L * 128 * 256);
  float* subX     = alloc(16L * 128 * 256);
  float* dfea     = alloc(16L * 128 * 256);
  float* encp     = alloc(16L * 128 * 256);
  float* decp     = alloc(16L * 128 * 256);
  float* qb       = alloc(16L * 128 * 256);
  float* wqb      = alloc(16L * 128 * 256);
  float* wk2b     = alloc(16L * 128 * 256);
  float* logits   = alloc(16L * 256 * 256);
  float* atto     = alloc(16L * 128 * 256);
  float* memb     = alloc(16L * 128 * 256);
  float* tgtb     = alloc(16L * 128 * 256);
  float* tgt2     = alloc(16L * 128 * 256);

  // ----- con_trans / dis_trans: points-as-channels MLPs -----
  // Layer-1 input is swapaxes(search_fea): X[b,c(point),l(chan)] strides (32768,1,256)
  auto trans_stack = [&](const int* pi, float* outT) {
    conv_layer(stream, P[pi[3]],  P[pi[0]],  P[pi[2]],  P[pi[1]],
               search_fea, 32768, 1, 256, t0, 256, 128, 256, 1, mean, var);
    conv_layer(stream, P[pi[7]],  P[pi[4]],  P[pi[6]],  P[pi[5]],
               t0, 32768, 128, 1, t1, 256, 128, 256, 1, mean, var);
    conv_layer(stream, P[pi[11]], P[pi[8]],  P[pi[10]], P[pi[9]],
               t1, 32768, 128, 1, t0, 256, 128, 256, 1, mean, var);
    conv_layer(stream, P[pi[13]], P[pi[12]], nullptr, nullptr,
               t0, 32768, 128, 1, tr4, 128, 128, 256, 0, mean, var);
    transpose12_kernel<<<(16 * 128 * 128 + 255) / 256, 256, 0, stream>>>(
        outT, tr4, 16, 128, 128);
  };
  { int ci[14] = {22,23,24,25, 26,27,28,29, 30,31,32,33, 34,35}; trans_stack(ci, confea); }
  { int di[14] = {50,51,52,53, 54,55,56,57, 58,59,60,61, 62,63}; trans_stack(di, disfea); }

  // ----- con_fea_fuse -----
  confuse_stats_kernel<<<8, 256, 0, stream>>>(confea, sbuf, nrmb, 16);
  confuse_sim_kernel<<<8, 256, 0, stream>>>(temp_fea, temp_xyz, sbuf, nrmb,
                                            maxcla, gath, 16);
  build_fusion_kernel<<<(16 * 288 * 128 + 255) / 256, 256, 0, stream>>>(
      fusX, maxcla, gath, confea, 16);
  // con_mlp (260->128->128->128, all BN+relu); layer-0 weights zero-padded to K=288
  slice_pad_w_kernel<<<(128 * 288 + 255) / 256, 256, 0, stream>>>(
      wp288, P[13], 128, 260, 288, 0);
  conv_layer(stream, wp288, P[10], P[12], P[11], fusX, 36864, 128, 1,
             cm0, 128, 128, 288, 1, mean, var);
  conv_layer(stream, P[17], P[14], P[16], P[15], cm0, 16384, 128, 1,
             cm1, 128, 128, 128, 1, mean, var);
  conv_layer(stream, P[21], P[18], P[20], P[19], cm1, 16384, 128, 1,
             cm0, 128, 128, 128, 1, mean, var);   // con_feature = cm0

  // ----- dis_fea_fuse: diff/disf synthesized inside the WMMA loader -----
  conv_layer(stream, P[67], P[64], P[66], P[65], nullptr, 0, 0, 0,
             big0, 128, 16384, 128, 2 /*leaky0.2*/, mean, var,
             1 /*diff*/, disfea, temp_fea);
  conv_layer(stream, P[69], P[68], nullptr, nullptr, big0, 2097152, 16384, 1,
             big1, 128, 16384, 128, 3 /*sigmoid*/, mean, var);   // dw
  // dis_mlp[0] (K=131): xyz channels -> per-(b,m,i) bias term; rest is K=128 GEMM
  slice_pad_w_kernel<<<(128 * 128 + 255) / 256, 256, 0, stream>>>(
      wdis, P[45], 128, 131, 128, 3);
  xyzterm_kernel<<<(16 * 128 * 128 + 255) / 256, 256, 0, stream>>>(
      xyzt, P[45], P[42], temp_xyz);
  conv_layer(stream, wdis, nullptr, P[44], P[43], nullptr, 0, 0, 0,
             big0, 128, 16384, 128, 1, mean, var,
             2 /*disf*/, big1, temp_fea, xyzt);
  conv_layer(stream, P[49], P[46], P[48], P[47], big0, 2097152, 16384, 1,
             big1, 128, 16384, 128, 1, mean, var);
  maxpool_i_kernel<<<1024, 256, 0, stream>>>(disfeat, big1, 16);

  // ----- fuse + final_mlp -----
  concat_kernel<<<2048, 256, 0, stream>>>(fusedcat, cm0, disfeat, 16);
  conv_layer(stream, P[85], P[82], P[84], P[83], fusedcat, 32768, 256, 1,
             fa, 128, 256, 128, 1, mean, var);
  conv_layer(stream, P[89], P[86], P[88], P[87], fa, 32768, 256, 1,
             fb, 128, 256, 128, 1, mean, var);
  conv_layer(stream, P[91], P[90], nullptr, nullptr, fb, 32768, 256, 1,
             fusedout, 128, 256, 128, 0, mean, var);

  // ----- sub_mlp on (search_fea - fused) -----
  axpy_kernel<<<2048, 256, 0, stream>>>(subX, search_fea, fusedout, 1, 524288L);
  conv_layer(stream, P[95],  P[92],  P[94],  P[93],  subX, 32768, 256, 1,
             fa, 128, 256, 128, 1, mean, var);
  conv_layer(stream, P[99],  P[96],  P[98],  P[97],  fa, 32768, 256, 1,
             fb, 128, 256, 128, 1, mean, var);
  conv_layer(stream, P[103], P[100], P[102], P[101], fb, 32768, 256, 1,
             fa, 128, 256, 128, 1, mean, var);
  conv_layer(stream, P[105], P[104], nullptr, nullptr, fa, 32768, 256, 1,
             dfea, 128, 256, 128, 0, mean, var);

  // ----- positional encodings (K=3 layer in VALU, then K=128 WMMA layer) -----
  posenc_l1_kernel<<<2048, 256, 0, stream>>>(fa, P[73], P[70], search_xyz);
  bn_block(stream, fa, 128, 256, P[72], P[71], 1, mean, var);
  conv_layer(stream, P[75], P[74], nullptr, nullptr, fa, 32768, 256, 1,
             encp, 128, 256, 128, 0, mean, var);
  posenc_l1_kernel<<<2048, 256, 0, stream>>>(fa, P[39], P[36], search_xyz);
  bn_block(stream, fa, 128, 256, P[38], P[37], 1, mean, var);
  conv_layer(stream, P[41], P[40], nullptr, nullptr, fa, 32768, 256, 1,
             decp, 128, 256, 128, 0, mean, var);

  // ----- encoder self-attention (q=k=v, shared WK => project once) -----
  axpy_kernel<<<2048, 256, 0, stream>>>(qb, dfea, encp, 0, 524288L);
  att_proj(stream, P[9], P[8], qb, wqb);
  att_core(stream, wqb, wqb, qb, dfea, memb, logits, atto);

  // ----- decoder self-attention -----
  axpy_kernel<<<2048, 256, 0, stream>>>(qb, fusedout, decp, 0, 524288L);
  att_proj(stream, P[7], P[6], qb, wqb);
  att_core(stream, wqb, wqb, qb, fusedout, tgtb, logits, atto);

  // ----- decoder cross-attention (q=tgt, k=v=mem) -----
  att_proj(stream, P[5], P[4], tgtb, wqb);
  att_proj(stream, P[5], P[4], memb, wk2b);
  att_core(stream, wqb, wk2b, memb, tgtb, tgt2, logits, atto);

  // ----- fea_layer -> d_out -----
  conv_layer(stream, P[79], P[76], P[78], P[77], tgt2, 32768, 256, 1,
             fa, 128, 256, 128, 1, mean, var);
  conv_layer(stream, P[81], P[80], nullptr, nullptr, fa, 32768, 256, 1,
             out, 128, 256, 128, 0, mean, var);
}
